// MPNN_20486994002071
// MI455X (gfx1250) — matrix operand order repaired and checked by reading, not compiled, not run on hardware
//
#include <hip/hip_runtime.h>
#include <hip/hip_bf16.h>

// ---------------------------------------------------------------------------
// MPNN on MI455X (gfx1250, wave32, WMMA).
//
// Key fusion: msg[e,o] = sum_i hs[e,i]*We[e,i,o], with
//   We[e,i,o] = sum_k q[e,k]*W2[k,i*32+o] + b2[i*32+o],  q = relu(eh@W1+b1)
// => msg row e = (q[e,:] (x) hs[e,:])  @  W2 reshaped [1024,32]  +  hs @ B2.
// The [E,1024]@[1024,32] GEMM runs on v_wmma_f32_16x16x32_bf16; the A rows
// (outer products) are synthesized in registers with v_pk_mul_bf16, so the
// 1.3GB We tensor is never written to memory. msg staging uses
// global_load_async_to_lds_b128 (ASYNCcnt).
// ---------------------------------------------------------------------------

typedef __attribute__((ext_vector_type(16))) __bf16 v16bf;
typedef __attribute__((ext_vector_type(8)))  float  v8f;

#define N_NODES 100000
#define N_EDGES 320000
#define DIM     32
#define NTILES_N (N_NODES / 16)   // 6250
#define NW 8                      // waves per 256-thread block

// Swizzled bf16 weight bank offsets (in bf16 elements), inside d_ws.
enum {
  OFF_ENC  = 0,       // enc_W   [32,32]  -> 2 frags
  OFF_PROJ = 1024,    // proj_W  [32,32]
  OFF_EENC = 2048,    // eenc_W  [16,32] zero-padded K
  OFF_ENW1 = 3072,    // en_W1   [32,32]
  OFF_W2   = 4096,    // en_W2 viewed [1024,32] -> 64 frags
  OFF_B2   = 36864,   // en_b2 viewed [32,32]
  OFF_WI   = 37888,   // gru_Wi  [32,96] -> 6 frags
  OFF_WH   = 40960,   // gru_Wh  [32,96]
  OFF_D1   = 44032,   // dec_W1..3 [32,32]
  OFF_D2   = 45056,
  OFF_D3   = 46080,
  SW_TOTAL = 47104
};

__device__ __forceinline__ void lds_fence() {
  asm volatile("s_wait_dscnt 0" ::: "memory");
}
__device__ __forceinline__ void async_fence() {
  asm volatile("s_wait_asynccnt 0" ::: "memory");
}

// Low 32 bits of the generic shared-aperture address == LDS byte offset
__device__ __forceinline__ unsigned lds_off(const void* p) {
  return (unsigned)(uintptr_t)p;
}

// Async 16B copy global -> LDS, tracked by ASYNCcnt.
__device__ __forceinline__ void async_b128(unsigned lds_byte_off, const void* g) {
  asm volatile("global_load_async_to_lds_b128 %0, %1, off"
               :: "v"(lds_byte_off), "v"(g) : "memory");
}

__device__ __forceinline__ v8f wmma_bf16(v16bf a, v16bf b, v8f c) {
  return __builtin_amdgcn_wmma_f32_16x16x32_bf16(false, a, false, b,
                                                 (short)0, c, false, false);
}

// B fragment: 16 contiguous bf16 per lane (pre-swizzled), 32B aligned load.
__device__ __forceinline__ v16bf b_frag(const __bf16* sw, int f, int lane) {
  return *(const v16bf*)(sw + ((size_t)f * 32 + lane) * 16);
}

// A fragment (bf16 16x32): lanes 0-15 hold M=lane, K = e(+8 for e>=8);
// lanes 16-31 hold M=lane-16, K shifted by +8.
__device__ __forceinline__ v16bf a_frag_lds(const float (*rows)[32], int lane) {
  const int M  = lane & 15;
  const int kb = (lane < 16) ? 0 : 8;
  v16bf a;
#pragma unroll
  for (int e = 0; e < 16; ++e) {
    int k = e + ((e < 8) ? 0 : 8) + kb;
    a[e] = (__bf16)rows[M][k];
  }
  return a;
}

__device__ __forceinline__ float sigmoid_(float x) {
  return 1.0f / (1.0f + __expf(-x));
}
__device__ __forceinline__ float tanh_(float x) {
  float e = __expf(-2.0f * x);
  return (1.0f - e) / (1.0f + e);
}

// ---------------------------------------------------------------------------
// Weight prep: row-major f32 [K,N] -> bf16 B-fragments (zero-pad K to 32).
// ---------------------------------------------------------------------------
__device__ void swz(__bf16* dst, const float* src, int K, int N,
                    int tid, int stride) {
  int Kc = (K + 31) / 32;
  int Nt = N / 16;
  int total = Kc * Nt * 512;
  for (int idx = tid; idx < total; idx += stride) {
    int f = idx >> 9, w = idx & 511;
    int lane = w >> 4, e = w & 15;
    int kc = f / Nt, nt = f % Nt;
    int k = kc * 32 + e + ((e < 8) ? 0 : 8) + ((lane < 16) ? 0 : 8);
    int n = nt * 16 + (lane & 15);
    float v = (k < K) ? src[k * N + n] : 0.0f;
    dst[idx] = (__bf16)v;
  }
}

__global__ __launch_bounds__(256)
void prep_weights_kernel(__bf16* sw, const float* enc_W, const float* proj_W,
                         const float* eenc_W, const float* en_W1,
                         const float* en_W2, const float* en_b2,
                         const float* gru_Wi, const float* gru_Wh,
                         const float* dW1, const float* dW2, const float* dW3) {
  int tid = blockIdx.x * blockDim.x + threadIdx.x;
  int stride = gridDim.x * blockDim.x;
  swz(sw + OFF_ENC,  enc_W,  32,   32, tid, stride);
  swz(sw + OFF_PROJ, proj_W, 32,   32, tid, stride);
  swz(sw + OFF_EENC, eenc_W, 16,   32, tid, stride);
  swz(sw + OFF_ENW1, en_W1,  32,   32, tid, stride);
  swz(sw + OFF_W2,   en_W2,  1024, 32, tid, stride);  // [32,1024] == [1024,32] flat
  swz(sw + OFF_B2,   en_b2,  32,   32, tid, stride);
  swz(sw + OFF_WI,   gru_Wi, 32,   96, tid, stride);
  swz(sw + OFF_WH,   gru_Wh, 32,   96, tid, stride);
  swz(sw + OFF_D1,   dW1,    32,   32, tid, stride);
  swz(sw + OFF_D2,   dW2,    32,   32, tid, stride);
  swz(sw + OFF_D3,   dW3,    32,   32, tid, stride);
}

// ---------------------------------------------------------------------------
// Node encoder: h = relu(relu(relu(emb[type]) @ enc_W + enc_b) @ proj_W + proj_b)
// ---------------------------------------------------------------------------
__global__ __launch_bounds__(256)
void node_encode_kernel(const int* nfeats, const float* emb, const __bf16* sw,
                        const float* enc_b, const float* proj_b, float* h) {
  __shared__ __align__(16) float sRow[NW][16][32];
  __shared__ int sTyp[NW][16];
  int wave = threadIdx.x >> 5, lane = threadIdx.x & 31;
  int tile = blockIdx.x * NW + wave;
  if (tile >= NTILES_N) return;
  int base = tile * 16;

  if (lane < 16) sTyp[wave][lane] = nfeats[base + lane];
  lds_fence();
#pragma unroll 4
  for (int r = 0; r < 16; ++r) {
    int ty = sTyp[wave][r];
    sRow[wave][r][lane] = fmaxf(emb[ty * 32 + lane], 0.0f);
  }
  lds_fence();

  const int n0 = lane & 15, mb = (lane < 16) ? 0 : 8;
  v16bf a = a_frag_lds(sRow[wave], lane);
  v8f c0 = {}, c1 = {};
  c0 = wmma_bf16(a, b_frag(sw + OFF_ENC, 0, lane), c0);
  c1 = wmma_bf16(a, b_frag(sw + OFF_ENC, 1, lane), c1);
#pragma unroll
  for (int v = 0; v < 8; ++v) {
    sRow[wave][mb + v][n0]      = fmaxf(c0[v] + enc_b[n0], 0.0f);
    sRow[wave][mb + v][n0 + 16] = fmaxf(c1[v] + enc_b[n0 + 16], 0.0f);
  }
  lds_fence();

  a = a_frag_lds(sRow[wave], lane);
  v8f d0 = {}, d1 = {};
  d0 = wmma_bf16(a, b_frag(sw + OFF_PROJ, 0, lane), d0);
  d1 = wmma_bf16(a, b_frag(sw + OFF_PROJ, 1, lane), d1);
#pragma unroll
  for (int v = 0; v < 8; ++v) {
    h[(base + mb + v) * 32 + n0]      = fmaxf(d0[v] + proj_b[n0], 0.0f);
    h[(base + mb + v) * 32 + n0 + 16] = fmaxf(d1[v] + proj_b[n0 + 16], 0.0f);
  }
}

// ---------------------------------------------------------------------------
// Edge encoder: q = relu((efeats @ eenc_W + eenc_b) @ en_W1 + en_b1)  (bf16)
// ---------------------------------------------------------------------------
__global__ __launch_bounds__(256)
void edge_encode_kernel(const float* efeats, const __bf16* sw,
                        const float* eenc_b, const float* en_b1, __bf16* q) {
  __shared__ __align__(16) float sEf[NW][16][16];
  __shared__ __align__(16) float sEh[NW][16][32];
  int wave = threadIdx.x >> 5, lane = threadIdx.x & 31;
  int tile = blockIdx.x * NW + wave;
  if (tile >= N_EDGES / 16) return;
  int base = tile * 16;

#pragma unroll
  for (int rr = 0; rr < 8; ++rr) {
    int r = rr * 2 + (lane >> 4);
    int c = lane & 15;
    sEf[wave][r][c] = efeats[(base + r) * 16 + c];
  }
  lds_fence();

  const int M = lane & 15, kb = (lane < 16) ? 0 : 8;
  const int n0 = lane & 15, mb = (lane < 16) ? 0 : 8;
  v16bf a;
#pragma unroll
  for (int e = 0; e < 16; ++e) {
    int k = e + ((e < 8) ? 0 : 8) + kb;
    a[e] = (k < 16) ? (__bf16)sEf[wave][M][k] : (__bf16)0.0f;  // zero-pad K
  }
  v8f c0 = {}, c1 = {};
  c0 = wmma_bf16(a, b_frag(sw + OFF_EENC, 0, lane), c0);
  c1 = wmma_bf16(a, b_frag(sw + OFF_EENC, 1, lane), c1);
#pragma unroll
  for (int v = 0; v < 8; ++v) {
    sEh[wave][mb + v][n0]      = c0[v] + eenc_b[n0];
    sEh[wave][mb + v][n0 + 16] = c1[v] + eenc_b[n0 + 16];
  }
  lds_fence();

  a = a_frag_lds(sEh[wave], lane);
  v8f d0 = {}, d1 = {};
  d0 = wmma_bf16(a, b_frag(sw + OFF_ENW1, 0, lane), d0);
  d1 = wmma_bf16(a, b_frag(sw + OFF_ENW1, 1, lane), d1);
#pragma unroll
  for (int v = 0; v < 8; ++v) {
    q[(base + mb + v) * 32 + n0]      = (__bf16)fmaxf(d0[v] + en_b1[n0], 0.0f);
    q[(base + mb + v) * 32 + n0 + 16] = (__bf16)fmaxf(d1[v] + en_b1[n0 + 16], 0.0f);
  }
}

__global__ __launch_bounds__(256)
void agg_init_kernel(float* agg, const float* conv_b) {
  int idx = blockIdx.x * blockDim.x + threadIdx.x;
  if (idx < N_NODES * DIM) agg[idx] = conv_b[idx & 31];
}

// ---------------------------------------------------------------------------
// Message pass: 32 edges per wave (two A-tiles share every B fragment).
// msg = (q (x) hs) @ W2r + hs @ B2; scatter-add into agg[dst].
// K=1024 -> 32 bf16 chunks; chunk c's A fragment = splat(q[m,c]) * hs[m,:]
// via v_pk_mul_bf16. q/h tiles staged with global_load_async_to_lds_b128.
// ---------------------------------------------------------------------------
__global__ __launch_bounds__(256)
void msg_kernel(const int* src, const int* dst, const float* h,
                const __bf16* q, const __bf16* sw, float* agg) {
  __shared__ __align__(16) float  sH[NW][32][32];
  __shared__ __align__(16) __bf16 sQ[NW][32][32];
  __shared__ int sIdx[NW][64];   // [0..31]=src, [32..63]=dst
  int wave = threadIdx.x >> 5, lane = threadIdx.x & 31;
  int tile = blockIdx.x * NW + wave;            // 32-edge tile
  if (tile >= N_EDGES / 32) return;
  int base = tile * 32;

  sIdx[wave][lane]      = src[base + lane];
  sIdx[wave][32 + lane] = dst[base + lane];
  lds_fence();

  // q tile: 32 rows x 64B = 2048B -> 4 sweeps of 32 lanes x 16B (async).
  {
    const char* qg = (const char*)(q + (size_t)base * 32);
    unsigned qb = lds_off(&sQ[wave][0][0]);
#pragma unroll
    for (int i = 0; i < 4; ++i) {
      unsigned off = (unsigned)(i * 32 + lane) * 16u;
      async_b128(qb + off, qg + off);
    }
  }
  // h gather: 32 rows x 128B; 8 lanes x 16B per row -> 8 sweeps (async).
#pragma unroll
  for (int i = 0; i < 8; ++i) {
    int slot = i * 32 + lane;            // 0..255
    int r = slot >> 3, p = slot & 7;     // row, 16B-chunk within row
    const float* gp = h + (size_t)sIdx[wave][r] * 32 + p * 4;
    async_b128(lds_off(&sH[wave][r][p * 4]), gp);
  }
  async_fence();
  lds_fence();

  const int M = lane & 15, kb = (lane < 16) ? 0 : 8;
  v16bf ah0, ah1;
#pragma unroll
  for (int e = 0; e < 16; ++e) {
    int k = e + ((e < 8) ? 0 : 8) + kb;
    ah0[e] = (__bf16)sH[wave][M][k];
    ah1[e] = (__bf16)sH[wave][16 + M][k];
  }

  v8f acc00, acc01, acc10, acc11;
  {
    v16bf b0 = b_frag(sw + OFF_B2, 0, lane);
    v16bf b1 = b_frag(sw + OFF_B2, 1, lane);
    v8f z = {};
    acc00 = wmma_bf16(ah0, b0, z);
    acc01 = wmma_bf16(ah0, b1, z);
    acc10 = wmma_bf16(ah1, b0, z);
    acc11 = wmma_bf16(ah1, b1, z);
  }

  const __bf16* w2 = sw + OFF_W2;
#pragma unroll 4
  for (int c = 0; c < 32; ++c) {
    v16bf b0 = b_frag(w2, c * 2 + 0, lane);
    v16bf b1 = b_frag(w2, c * 2 + 1, lane);
    __bf16 q0 = sQ[wave][M][c];
    __bf16 q1 = sQ[wave][16 + M][c];
    v16bf qs0, qs1;
#pragma unroll
    for (int e = 0; e < 16; ++e) { qs0[e] = q0; qs1[e] = q1; }
    v16bf a0 = ah0 * qs0;            // v_pk_mul_bf16 x8
    v16bf a1 = ah1 * qs1;
    acc00 = wmma_bf16(a0, b0, acc00);
    acc01 = wmma_bf16(a0, b1, acc01);
    acc10 = wmma_bf16(a1, b0, acc10);
    acc11 = wmma_bf16(a1, b1, acc11);
  }

  const int n0 = lane & 15, mb = (lane < 16) ? 0 : 8;
#pragma unroll
  for (int v = 0; v < 8; ++v) {
    int d0i = sIdx[wave][32 + mb + v];
    int d1i = sIdx[wave][48 + mb + v];
    atomicAdd(&agg[d0i * 32 + n0],      acc00[v]);
    atomicAdd(&agg[d0i * 32 + n0 + 16], acc01[v]);
    atomicAdd(&agg[d1i * 32 + n0],      acc10[v]);
    atomicAdd(&agg[d1i * 32 + n0 + 16], acc11[v]);
  }
}

// ---------------------------------------------------------------------------
// GRU step: x=relu(agg); gi = x@Wi+bi; gh = h@Wh+bh; gates; h update in place.
// ---------------------------------------------------------------------------
__global__ __launch_bounds__(256)
void gru_kernel(const float* agg, float* h, const __bf16* sw,
                const float* gru_bi, const float* gru_bh) {
  __shared__ __align__(16) float sX[NW][16][32];
  __shared__ __align__(16) float sHo[NW][16][32];
  int wave = threadIdx.x >> 5, lane = threadIdx.x & 31;
  int tile = blockIdx.x * NW + wave;
  if (tile >= NTILES_N) return;
  int base = tile * 16;

#pragma unroll 4
  for (int r = 0; r < 16; ++r) {
    sX[wave][r][lane]  = fmaxf(agg[(base + r) * 32 + lane], 0.0f);
    sHo[wave][r][lane] = h[(base + r) * 32 + lane];
  }
  lds_fence();

  v16bf ax = a_frag_lds(sX[wave], lane);
  v16bf ah = a_frag_lds(sHo[wave], lane);
  v8f gi[6], gh[6];
#pragma unroll
  for (int t = 0; t < 6; ++t) {
    v8f z0 = {}, z1 = {};
    gi[t] = wmma_bf16(ax, b_frag(sw + OFF_WI, t, lane), z0);
    gh[t] = wmma_bf16(ah, b_frag(sw + OFF_WH, t, lane), z1);
  }

  const int n0 = lane & 15, mb = (lane < 16) ? 0 : 8;
#pragma unroll
  for (int tc = 0; tc < 2; ++tc) {
    int d = n0 + 16 * tc;
    float bir = gru_bi[d], biz = gru_bi[32 + d], bin = gru_bi[64 + d];
    float bhr = gru_bh[d], bhz = gru_bh[32 + d], bhn = gru_bh[64 + d];
#pragma unroll
    for (int v = 0; v < 8; ++v) {
      int m = mb + v;
      float ir = gi[tc][v] + bir, iz = gi[2 + tc][v] + biz, in_ = gi[4 + tc][v] + bin;
      float hr = gh[tc][v] + bhr, hz = gh[2 + tc][v] + bhz, hn = gh[4 + tc][v] + bhn;
      float r  = sigmoid_(ir + hr);
      float z  = sigmoid_(iz + hz);
      float ng = tanh_(in_ + r * hn);
      float ho = sHo[wave][m][d];
      h[(base + m) * 32 + d] = (1.0f - z) * ng + z * ho;
    }
  }
}

// ---------------------------------------------------------------------------
// Decoder: 3x (Linear 32x32 + PReLU) via WMMA chained through LDS, then
// tiny [32,3] tail on VALU.
// ---------------------------------------------------------------------------
__global__ __launch_bounds__(256)
void decoder_kernel(const float* h, const __bf16* sw,
                    const float* b1, const float* a1,
                    const float* b2, const float* a2,
                    const float* b3, const float* a3,
                    const float* W4, const float* b4, float* out) {
  __shared__ __align__(16) float sY[NW][16][32];
  int wave = threadIdx.x >> 5, lane = threadIdx.x & 31;
  int tile = blockIdx.x * NW + wave;
  if (tile >= NTILES_N) return;
  int base = tile * 16;

#pragma unroll 4
  for (int r = 0; r < 16; ++r) sY[wave][r][lane] = h[(base + r) * 32 + lane];
  lds_fence();

  const int n0 = lane & 15, mb = (lane < 16) ? 0 : 8;
  const int offs[3] = {OFF_D1, OFF_D2, OFF_D3};
  const float* bls[3] = {b1, b2, b3};
  const float* als[3] = {a1, a2, a3};

#pragma unroll
  for (int l = 0; l < 3; ++l) {
    v16bf a = a_frag_lds(sY[wave], lane);
    v8f c0 = {}, c1 = {};
    c0 = wmma_bf16(a, b_frag(sw + offs[l], 0, lane), c0);
    c1 = wmma_bf16(a, b_frag(sw + offs[l], 1, lane), c1);
    float al = als[l][0];
    const float* bl = bls[l];
#pragma unroll
    for (int v = 0; v < 8; ++v) {
      float y0 = c0[v] + bl[n0];
      float y1 = c1[v] + bl[n0 + 16];
      sY[wave][mb + v][n0]      = (y0 >= 0.0f) ? y0 : al * y0;
      sY[wave][mb + v][n0 + 16] = (y1 >= 0.0f) ? y1 : al * y1;
    }
    lds_fence();
  }

  if (lane < 16) {
#pragma unroll
    for (int o = 0; o < 3; ++o) {
      float acc = b4[o];
#pragma unroll 8
      for (int d = 0; d < 32; ++d) acc += sY[wave][lane][d] * W4[d * 3 + o];
      out[(base + lane) * 3 + o] = acc;
    }
  }
}

// ---------------------------------------------------------------------------
extern "C" void kernel_launch(void* const* d_in, const int* in_sizes, int n_in,
                              void* d_out, int out_size, void* d_ws, size_t ws_size,
                              hipStream_t stream) {
  (void)in_sizes; (void)n_in; (void)out_size; (void)ws_size;
  const int*   nfeats = (const int*)d_in[0];
  const float* efeats = (const float*)d_in[1];
  const int*   src    = (const int*)d_in[2];
  const int*   dst    = (const int*)d_in[3];
  const float* emb    = (const float*)d_in[4];
  const float* enc_W  = (const float*)d_in[5];
  const float* enc_b  = (const float*)d_in[6];
  const float* eenc_W = (const float*)d_in[7];
  const float* eenc_b = (const float*)d_in[8];
  const float* proj_W = (const float*)d_in[9];
  const float* proj_b = (const float*)d_in[10];
  const float* en_W1  = (const float*)d_in[11];
  const float* en_b1  = (const float*)d_in[12];
  const float* en_W2  = (const float*)d_in[13];
  const float* en_b2  = (const float*)d_in[14];
  const float* conv_b = (const float*)d_in[15];
  const float* gru_Wi = (const float*)d_in[16];
  const float* gru_Wh = (const float*)d_in[17];
  const float* gru_bi = (const float*)d_in[18];
  const float* gru_bh = (const float*)d_in[19];
  const float* dW1 = (const float*)d_in[20];
  const float* db1 = (const float*)d_in[21];
  const float* da1 = (const float*)d_in[22];
  const float* dW2 = (const float*)d_in[23];
  const float* db2 = (const float*)d_in[24];
  const float* da2 = (const float*)d_in[25];
  const float* dW3 = (const float*)d_in[26];
  const float* db3 = (const float*)d_in[27];
  const float* da3 = (const float*)d_in[28];
  const float* dW4 = (const float*)d_in[29];
  const float* db4 = (const float*)d_in[30];

  // Workspace layout: h | agg | q(bf16) | swizzled weights(bf16). ~46.2 MB.
  const size_t H_BYTES = (size_t)N_NODES * DIM * sizeof(float);   // 12.8 MB
  const size_t Q_BYTES = (size_t)N_EDGES * DIM * sizeof(__bf16);  // 20.5 MB
  char*   ws  = (char*)d_ws;
  float*  h   = (float*)ws;
  float*  agg = (float*)(ws + H_BYTES);
  __bf16* q   = (__bf16*)(ws + 2 * H_BYTES);
  __bf16* sw  = (__bf16*)(ws + 2 * H_BYTES + Q_BYTES);

  prep_weights_kernel<<<64, 256, 0, stream>>>(sw, enc_W, proj_W, eenc_W, en_W1,
                                              en_W2, en_b2, gru_Wi, gru_Wh,
                                              dW1, dW2, dW3);

  const int nblk  = (NTILES_N + NW - 1) / NW;        // 782
  const int eblk  = (N_EDGES / 16 + NW - 1) / NW;    // 2500 (16-edge tiles)
  const int eblk2 = (N_EDGES / 32 + NW - 1) / NW;    // 1250 (32-edge tiles)

  node_encode_kernel<<<nblk, 256, 0, stream>>>(nfeats, emb, sw, enc_b, proj_b, h);
  edge_encode_kernel<<<eblk, 256, 0, stream>>>(efeats, sw, eenc_b, en_b1, q);

  for (int step = 0; step < 3; ++step) {
    agg_init_kernel<<<(N_NODES * DIM + 255) / 256, 256, 0, stream>>>(agg, conv_b);
    msg_kernel<<<eblk2, 256, 0, stream>>>(src, dst, h, q, sw, agg);
    gru_kernel<<<nblk, 256, 0, stream>>>(agg, h, sw, gru_bi, gru_bh);
  }

  decoder_kernel<<<nblk, 256, 0, stream>>>(h, sw, db1, da1, db2, da2, db3, da3,
                                           dW4, db4, (float*)d_out);
}
